// GraphSAGEEncoder_71098888618518
// MI455X (gfx1250) — matrix operand order repaired and checked by reading, not compiled, not run on hardware
//
#include <hip/hip_runtime.h>
#include <hip/hip_bf16.h>

#define N_NODES 100000
#define E_EDGES 1600000
#define B_GR    64
#define D_IN    9
#define H       128
#define LN_EPS  1e-5f

typedef __attribute__((ext_vector_type(16))) _Float16 v16h;
typedef __attribute__((ext_vector_type(8)))  _Float16 v8h;
typedef __attribute__((ext_vector_type(8)))  float    v8f;
typedef __attribute__((ext_vector_type(4)))  float    v4f;
typedef __attribute__((ext_vector_type(4)))  int      v4i;

#ifndef __has_builtin
#define __has_builtin(x) 0
#endif
#if __has_builtin(__builtin_amdgcn_global_load_async_to_lds_b128)
#define HAVE_ASYNC_LDS 1
#else
#define HAVE_ASYNC_LDS 0
#endif

typedef __attribute__((address_space(1))) v4i gv4i;   // global int4
typedef __attribute__((address_space(3))) v4i lv4i;   // LDS int4

// ---------------------------------------------------------------- utilities
__global__ void k_zero_f32(float* __restrict__ p, long n) {
    long i = (long)blockIdx.x * blockDim.x + threadIdx.x;
    long stride = (long)gridDim.x * blockDim.x;
    for (; i < n; i += stride) p[i] = 0.0f;
}

__global__ void k_cvt_f16(const float* __restrict__ w, _Float16* __restrict__ o, int n) {
    int i = blockIdx.x * blockDim.x + threadIdx.x;
    int stride = gridDim.x * blockDim.x;
    for (; i < n; i += stride) o[i] = (_Float16)w[i];
}

// scale summed aggregate by 1/max(deg,1) and convert to f16 (WMMA A operand feed)
__global__ __launch_bounds__(256)
void k_scale_cvt(const float* __restrict__ agg, const float* __restrict__ deg,
                 _Float16* __restrict__ aggh) {
    long t = (long)blockIdx.x * blockDim.x + threadIdx.x;
    if (t >= (long)N_NODES * H / 4) return;
    int n = (int)(t >> 5);                       // 32 chunks of 4 per row
    float inv = 1.0f / fmaxf(deg[n], 1.0f);
    v4f a = *(const v4f*)(agg + t * 4);
    _Float16* o = aggh + t * 4;
    o[0] = (_Float16)(a.x * inv);
    o[1] = (_Float16)(a.y * inv);
    o[2] = (_Float16)(a.z * inv);
    o[3] = (_Float16)(a.w * inv);
}

// ------------------------------------------------ node embedder: Lin(9->128)+LN+ReLU
// one wave (32 lanes) per node; 4 channels per lane; LN via full-wave shfl_xor
__global__ __launch_bounds__(256)
void k_embed(const float* __restrict__ x,
             const float* __restrict__ W0, const float* __restrict__ b0,
             const float* __restrict__ g0, const float* __restrict__ be0,
             _Float16* __restrict__ h) {
    int node = (blockIdx.x * blockDim.x + threadIdx.x) >> 5;
    int lane = threadIdx.x & 31;
    if (node >= N_NODES) return;

    float xv[D_IN];
#pragma unroll
    for (int i = 0; i < D_IN; ++i) xv[i] = x[(size_t)node * D_IN + i];

    float o[4];
#pragma unroll
    for (int j = 0; j < 4; ++j) {
        int c = j * 32 + lane;
        const float* wr = W0 + (size_t)c * D_IN;
        float a = b0[c];
#pragma unroll
        for (int i = 0; i < D_IN; ++i) a += wr[i] * xv[i];
        o[j] = a;
    }
    float s1 = o[0] + o[1] + o[2] + o[3];
    float s2 = o[0]*o[0] + o[1]*o[1] + o[2]*o[2] + o[3]*o[3];
#pragma unroll
    for (int m = 1; m < 32; m <<= 1) {
        s1 += __shfl_xor(s1, m, 32);
        s2 += __shfl_xor(s2, m, 32);
    }
    float mu = s1 * (1.0f / H);
    float var = s2 * (1.0f / H) - mu * mu;
    float rs = rsqrtf(var + LN_EPS);
#pragma unroll
    for (int j = 0; j < 4; ++j) {
        int c = j * 32 + lane;
        float v = (o[j] - mu) * rs * g0[c] + be0[c];
        v = fmaxf(v, 0.0f);
        h[(size_t)node * H + c] = (_Float16)v;
    }
}

// ------------------------------------------------------------ degree count
__global__ void k_deg(const int* __restrict__ ei, float* __restrict__ deg) {
    int e = blockIdx.x * blockDim.x + threadIdx.x;
    if (e < E_EDGES) atomicAdd(&deg[ei[E_EDGES + e]], 1.0f);
}

// ---------------------------------------------- scatter: agg[dst] += h[src]
// thread handles one (edge, 32-channel chunk); f16 gather, f32 atomic adds
__global__ __launch_bounds__(256)
void k_scatter(const _Float16* __restrict__ h, const int* __restrict__ ei,
               float* __restrict__ agg) {
    long t = (long)blockIdx.x * blockDim.x + threadIdx.x;
    if (t >= (long)E_EDGES * 4) return;
    int e = (int)(t >> 2);
    int q = (int)(t & 3);
    int s = ei[e];
    int d = ei[E_EDGES + e];
    const _Float16* hp = h + (size_t)s * H + q * 32;
    float* ap = agg + (size_t)d * H + q * 32;
#pragma unroll
    for (int i = 0; i < 4; ++i) {
        v8h hv = *(const v8h*)(hp + i * 8);
#pragma unroll
        for (int j = 0; j < 8; ++j) atomicAdd(ap + i * 8 + j, (float)hv[j]);
    }
}

// ------------------- fused SAGE update: out = relu(LN( aggh@Wl^T + bl + h@Wr^T ))
// one wave -> 16 rows x 128 cols; 64x v_wmma_f32_16x16x32_f16 per wave.
// Weights staged to LDS via GLOBAL_LOAD_ASYNC_TO_LDS_B128 when available.
__global__ __launch_bounds__(256)
void k_sage_gemm(const _Float16* __restrict__ aggh,   // (agg/deg) as f16
                 const _Float16* __restrict__ h,      // root input f16
                 const _Float16* __restrict__ Wl16,   // [H][H] out-major (n,k)
                 const _Float16* __restrict__ Wr16,
                 const float* __restrict__ bl,
                 const float* __restrict__ gamma, const float* __restrict__ beta,
                 _Float16* __restrict__ out16,        // one of out16/out32 is used
                 float* __restrict__ out32) {
    __shared__ __align__(32) _Float16 sWl[H * H];  // 32 KB
    __shared__ __align__(32) _Float16 sWr[H * H];  // 32 KB

    int tid = threadIdx.x;
    // ---- stage both weight matrices into LDS ----
#if HAVE_ASYNC_LDS
    {
        // v4i-typed pointers in explicit address spaces (prototype: vV4i*1 V4i*3 Ii Ii)
        gv4i* gl = (gv4i*)(unsigned long long)(size_t)Wl16;
        gv4i* gr = (gv4i*)(unsigned long long)(size_t)Wr16;
        lv4i* ll = (lv4i*)(unsigned int)(size_t)&sWl[0];
        lv4i* lr = (lv4i*)(unsigned int)(size_t)&sWr[0];
        // 2048 x 16B chunks per matrix, cooperatively issued, ASYNCcnt-tracked
        for (int i = tid; i < H * H / 8; i += 256) {
            __builtin_amdgcn_global_load_async_to_lds_b128(gl + i, ll + i, 0, 0);
            __builtin_amdgcn_global_load_async_to_lds_b128(gr + i, lr + i, 0, 0);
        }
#if __has_builtin(__builtin_amdgcn_s_wait_asynccnt)
        __builtin_amdgcn_s_wait_asynccnt(0);
#else
        asm volatile("s_wait_asynccnt 0x0" ::: "memory");
#endif
    }
#else
    {
        const uint4* sl = (const uint4*)Wl16;
        const uint4* sr = (const uint4*)Wr16;
        uint4* dl = (uint4*)sWl;
        uint4* dr = (uint4*)sWr;
        for (int i = tid; i < H * H / 8; i += 256) { dl[i] = sl[i]; dr[i] = sr[i]; }
    }
#endif
    __syncthreads();

    int lane = tid & 31;
    int wave = tid >> 5;
    long nbase = ((long)blockIdx.x * 8 + wave) * 16;
    if (nbase >= N_NODES) return;   // wave-uniform guard -> EXEC all-ones below

    int row  = (int)nbase + (lane & 15);
    int half = lane >> 4;           // 0: lanes 0-15, 1: lanes 16-31

    // Build A fragments in contiguous k order (this code's choice, same as B):
    //   lane l holds row M=l&15; halves [0..15] = K=32s+16*half .. +15
    //   -> one contiguous 32B run (two 16B loads), matching B
    union AV { v16h v; v8h p[2]; };
    v16h Aagg[4], Aroot[4];
#pragma unroll
    for (int s = 0; s < 4; ++s) {
        int cb = 32 * s + half * 16;
        AV ua, ur;
        const _Float16* ap = aggh + (size_t)row * H + cb;
        ua.p[0] = *(const v8h*)(ap + 0);
        ua.p[1] = *(const v8h*)(ap + 8);
        Aagg[s] = ua.v;
        const _Float16* hp = h + (size_t)row * H + cb;
        ur.p[0] = *(const v8h*)(hp + 0);
        ur.p[1] = *(const v8h*)(hp + 8);
        Aroot[s] = ur.v;
    }

    // accumulate both GEMMs into the same 16x128 accumulator tile
    v8f acc[8];
#pragma unroll
    for (int nt = 0; nt < 8; ++nt) acc[nt] = (v8f){0,0,0,0,0,0,0,0};

    int ncol = (lane & 15);
#pragma unroll
    for (int nt = 0; nt < 8; ++nt) {
        int n = nt * 16 + ncol;
#pragma unroll
        for (int s = 0; s < 4; ++s) {
            // B frag: lane l holds col N=l&15, contiguous K-run of 16 at 32s+16*half
            v16h b = *(const v16h*)&sWl[(size_t)n * H + 32 * s + half * 16];
            acc[nt] = __builtin_amdgcn_wmma_f32_16x16x32_f16(
                false, Aagg[s], false, b, (short)0, acc[nt], false, false);
        }
#pragma unroll
        for (int s = 0; s < 4; ++s) {
            v16h b = *(const v16h*)&sWr[(size_t)n * H + 32 * s + half * 16];
            acc[nt] = __builtin_amdgcn_wmma_f32_16x16x32_f16(
                false, Aroot[s], false, b, (short)0, acc[nt], false, false);
        }
    }

    // bias (lin_l), per-row LayerNorm, ReLU, store.
    // C/D layout: element e of lane l -> row M = e + 8*half, col N = nt*16 + (l&15)
#pragma unroll
    for (int nt = 0; nt < 8; ++nt) {
        float blv = bl[nt * 16 + ncol];
#pragma unroll
        for (int e = 0; e < 8; ++e) acc[nt][e] += blv;
    }
    float s1[8], s2[8];
#pragma unroll
    for (int e = 0; e < 8; ++e) { s1[e] = 0.0f; s2[e] = 0.0f; }
#pragma unroll
    for (int nt = 0; nt < 8; ++nt)
#pragma unroll
        for (int e = 0; e < 8; ++e) {
            float v = acc[nt][e];
            s1[e] += v; s2[e] += v * v;
        }
    // reduce each row over its 16-lane half-wave (masks 1,2,4,8 stay in-half)
#pragma unroll
    for (int m = 1; m < 16; m <<= 1)
#pragma unroll
        for (int e = 0; e < 8; ++e) {
            s1[e] += __shfl_xor(s1[e], m, 32);
            s2[e] += __shfl_xor(s2[e], m, 32);
        }
    float mu[8], rs[8];
#pragma unroll
    for (int e = 0; e < 8; ++e) {
        mu[e] = s1[e] * (1.0f / H);
        float var = s2[e] * (1.0f / H) - mu[e] * mu[e];
        rs[e] = rsqrtf(var + LN_EPS);
    }
#pragma unroll
    for (int nt = 0; nt < 8; ++nt) {
        int col = nt * 16 + ncol;
        float gv = gamma[col], bv = beta[col];
#pragma unroll
        for (int e = 0; e < 8; ++e) {
            float v = (acc[nt][e] - mu[e]) * rs[e] * gv + bv;
            v = fmaxf(v, 0.0f);
            size_t idx = (size_t)(nbase + e + 8 * half) * H + col;
            if (out32) out32[idx] = v;
            else       out16[idx] = (_Float16)v;
        }
    }
}

// ------------------------------------------------------------- graph pooling
__global__ void k_pool_cnt(const int* __restrict__ batch, float* __restrict__ cnt) {
    int n = blockIdx.x * blockDim.x + threadIdx.x;
    if (n < N_NODES) atomicAdd(&cnt[batch[n]], 1.0f);
}

__global__ __launch_bounds__(256)
void k_pool_acc(const float* __restrict__ ne, const int* __restrict__ batch,
                float* __restrict__ gmean, float* __restrict__ gmax) {
    long t = (long)blockIdx.x * blockDim.x + threadIdx.x;
    if (t >= (long)N_NODES * H) return;
    int n = (int)(t >> 7);
    int c = (int)(t & (H - 1));
    int g = batch[n];
    float v = ne[t];
    atomicAdd(&gmean[g * H + c], v);
    atomicMax((int*)&gmax[g * H + c], __float_as_int(v));  // v >= 0 after ReLU
}

__global__ void k_pool_fin(const float* __restrict__ gmean, const float* __restrict__ gmax,
                           const float* __restrict__ cnt, float* __restrict__ out) {
    int t = blockIdx.x * blockDim.x + threadIdx.x;
    if (t >= B_GR * H) return;
    int g = t >> 7, c = t & (H - 1);
    float cm = fmaxf(cnt[g], 1.0f);
    out[(size_t)g * 2 * H + c]     = gmean[t] / cm;
    out[(size_t)g * 2 * H + H + c] = gmax[t];
}

// ---------------------------------------------------------------- launcher
extern "C" void kernel_launch(void* const* d_in, const int* in_sizes, int n_in,
                              void* d_out, int out_size, void* d_ws, size_t ws_size,
                              hipStream_t stream) {
    const float* x   = (const float*)d_in[0];
    const int*   ei  = (const int*)d_in[1];
    const int*   bat = (const int*)d_in[2];
    const float* W0  = (const float*)d_in[3];
    const float* b0  = (const float*)d_in[4];
    const float* g0  = (const float*)d_in[5];
    const float* be0 = (const float*)d_in[6];
    const float* Wl[3] = {(const float*)d_in[7],  (const float*)d_in[12], (const float*)d_in[17]};
    const float* bl[3] = {(const float*)d_in[8],  (const float*)d_in[13], (const float*)d_in[18]};
    const float* Wr[3] = {(const float*)d_in[9],  (const float*)d_in[14], (const float*)d_in[19]};
    const float* gg[3] = {(const float*)d_in[10], (const float*)d_in[15], (const float*)d_in[20]};
    const float* bb[3] = {(const float*)d_in[11], (const float*)d_in[16], (const float*)d_in[21]};

    // workspace layout (256B aligned chunks)
    char* base = (char*)d_ws;
    size_t off = 0;
    auto take = [&](size_t bytes) -> void* {
        void* r = base + off;
        off += (bytes + 255) & ~(size_t)255;
        return r;
    };
    _Float16* h     = (_Float16*)take((size_t)N_NODES * H * sizeof(_Float16));
    float*    agg   = (float*)take((size_t)N_NODES * H * sizeof(float));
    _Float16* aggh  = (_Float16*)take((size_t)N_NODES * H * sizeof(_Float16));
    float*    deg   = (float*)take((size_t)N_NODES * sizeof(float));
    _Float16* w16   = (_Float16*)take((size_t)6 * H * H * sizeof(_Float16));
    float*    gmean = (float*)take((size_t)B_GR * H * sizeof(float));
    float*    gmax  = (float*)take((size_t)B_GR * H * sizeof(float));
    float*    gcnt  = (float*)take((size_t)B_GR * sizeof(float));
    (void)ws_size; (void)in_sizes; (void)n_in; (void)out_size;

    float* nodeEmbed  = (float*)d_out;                        // N*H
    float* graphEmbed = (float*)d_out + (size_t)N_NODES * H;  // B*2H

    // convert all SAGE weights to f16 (out-major [n][k] = row-major as stored)
    for (int l = 0; l < 3; ++l) {
        k_cvt_f16<<<64, 256, 0, stream>>>(Wl[l], w16 + (size_t)(2 * l)     * H * H, H * H);
        k_cvt_f16<<<64, 256, 0, stream>>>(Wr[l], w16 + (size_t)(2 * l + 1) * H * H, H * H);
    }

    // node embedder
    k_embed<<<(N_NODES * 32 + 255) / 256, 256, 0, stream>>>(x, W0, b0, g0, be0, h);

    // degrees (same for every layer)
    k_zero_f32<<<1024, 256, 0, stream>>>(deg, N_NODES);
    k_deg<<<(E_EDGES + 255) / 256, 256, 0, stream>>>(ei, deg);

    const long aggN = (long)N_NODES * H;
    const int gemmBlocks = (N_NODES / 16 + 7) / 8;   // 8 waves/block, 16 rows/wave

    for (int l = 0; l < 3; ++l) {
        k_zero_f32<<<8192, 256, 0, stream>>>(agg, aggN);
        k_scatter<<<(int)(((long)E_EDGES * 4 + 255) / 256), 256, 0, stream>>>(h, ei, agg);
        k_scale_cvt<<<(int)((aggN / 4 + 255) / 256), 256, 0, stream>>>(agg, deg, aggh);
        bool last = (l == 2);
        k_sage_gemm<<<gemmBlocks, 256, 0, stream>>>(
            aggh, h,
            w16 + (size_t)(2 * l) * H * H, w16 + (size_t)(2 * l + 1) * H * H,
            bl[l], gg[l], bb[l],
            last ? nullptr : h,
            last ? nodeEmbed : nullptr);
    }

    // pooling
    k_zero_f32<<<64, 256, 0, stream>>>(gmean, B_GR * H);
    k_zero_f32<<<64, 256, 0, stream>>>(gmax, B_GR * H);
    k_zero_f32<<<1, 256, 0, stream>>>(gcnt, B_GR);
    k_pool_cnt<<<(N_NODES + 255) / 256, 256, 0, stream>>>(bat, gcnt);
    k_pool_acc<<<(int)((aggN + 255) / 256), 256, 0, stream>>>(nodeEmbed, bat, gmean, gmax);
    k_pool_fin<<<(B_GR * H + 255) / 256, 256, 0, stream>>>(gmean, gmax, gcnt, graphEmbed);
}
